// OneToManyMultiheadMonotonicAttention_6090263626211
// MI455X (gfx1250) — matrix operand order repaired and checked by reading, not compiled, not run on hardware
//
#include <hip/hip_runtime.h>

typedef __attribute__((ext_vector_type(16))) __bf16 v16bf;
typedef __attribute__((ext_vector_type(8)))  __bf16 v8bf;
typedef __attribute__((ext_vector_type(8)))  float  v8f;
typedef __attribute__((ext_vector_type(4)))  float  v4f;
typedef __attribute__((ext_vector_type(4)))  unsigned int u32x4;
typedef __attribute__((ext_vector_type(8)))  unsigned int u32x8;

#define NEG_INF   (-1e30f)
#define CLAMP_PHI 3.8918202981106265f   /* -ln(1/0.98 - 1) = ln(49) */
#define SCALE_QK  0.125f                /* 1/sqrt(64) */

#define USE_TDM 1   /* stage K tile with tensor_load_to_lds (TDM) */

// ---------------------------------------------------------------------------
// WMMA fragment helpers (16x16x32 bf16; A/B use this code's contiguous k order, not AMD's table order):
//  A (16xK=32): lane%16 = row M; lane/16 selects K 0..15 vs 16..31 (code's order)
//               -> one contiguous 16xbf16 run (two 16B loads).
//  B (K=32x16): lane%16 = col N; lane/16 selects K 0..15 vs 16..31 (code's order)
//               -> one contiguous 16xbf16 run (two 16B loads).
//  C/D f32:     VGPR r -> M = r + (lane/16)*8, N = lane%16.
// ---------------------------------------------------------------------------
__device__ __forceinline__ v16bf frag_a(const __bf16* base, int stride) {
  int lane = threadIdx.x & 31;
  const __bf16* p = base + (lane & 15) * stride + ((lane >> 4) << 4);
  v8bf lo = *(const v8bf*)p;
  v8bf hi = *(const v8bf*)(p + 8);
  return __builtin_shufflevector(lo, hi, 0,1,2,3,4,5,6,7,8,9,10,11,12,13,14,15);
}
__device__ __forceinline__ v16bf frag_b(const __bf16* base, int stride) {
  int lane = threadIdx.x & 31;
  const __bf16* p = base + (lane & 15) * stride + ((lane >> 4) << 4);
  v8bf lo = *(const v8bf*)p;
  v8bf hi = *(const v8bf*)(p + 8);
  return __builtin_shufflevector(lo, hi, 0,1,2,3,4,5,6,7,8,9,10,11,12,13,14,15);
}
__device__ __forceinline__ v8f wmma_bf16(v16bf a, v16bf b, v8f c) {
  return __builtin_amdgcn_wmma_f32_16x16x32_bf16(false, a, false, b, (short)0, c,
                                                 false, false);
}
__device__ __forceinline__ float log_sigmoid(float x) {
  return fminf(x, 0.f) - log1pf(__expf(-fabsf(x)));
}
__device__ __forceinline__ float logaddexp_f(float a, float b) {
  float m = fmaxf(a, b);
  return m + log1pf(__expf(-fabsf(a - b)));
}

// ---------------------------------------------------------------------------
// Generic projection GEMM: C = A[MxK] * W[NxK]^T + bias[N], bf16 WMMA inside.
// Block 256 thr (8 waves), tile 128x64, K-chunks of 32 staged via LDS.
// mode 0: f32 row-major C (ldc);  mode 1: bf16 [b][h][k][d];  mode 2: bf16 [b][h][d][k]
// ---------------------------------------------------------------------------
#define SA_STR 40
#define SB_STR 40

__global__ __launch_bounds__(256) void mono_proj_gemm(
    const float* __restrict__ A, const float* __restrict__ W,
    const float* __restrict__ bias, void* __restrict__ C,
    int M, int N, int K, int mode, int ldc) {
  __shared__ __bf16 sA[128 * SA_STR];
  __shared__ __bf16 sB[64 * SB_STR];
  const int tid  = threadIdx.x;
  const int lane = tid & 31;
  const int wave = tid >> 5;
  const int wm = wave >> 1, wn = wave & 1;
  const int m0 = blockIdx.y * 128, n0 = blockIdx.x * 64;

  // precomputed staging coordinates (fixed trip counts -> batched loads)
  const int smR = tid >> 3;            // A tile row for chunk i=0 (+32/iter)
  const int skk = (tid & 7) << 2;      // k offset within 32-chunk

  v8f acc[2][2];
  for (int i = 0; i < 2; ++i)
    for (int j = 0; j < 2; ++j)
      for (int r = 0; r < 8; ++r) acc[i][j][r] = 0.f;

  for (int k0 = 0; k0 < K; k0 += 32) {
    v4f ta[4], tb[2];
#pragma unroll
    for (int i = 0; i < 4; ++i)
      ta[i] = *(const v4f*)(A + (size_t)(m0 + smR + i * 32) * K + k0 + skk);
#pragma unroll
    for (int i = 0; i < 2; ++i)
      tb[i] = *(const v4f*)(W + (size_t)(n0 + smR + i * 32) * K + k0 + skk);
#pragma unroll
    for (int i = 0; i < 4; ++i) {
      __bf16* d = sA + (smR + i * 32) * SA_STR + skk;
      d[0] = (__bf16)ta[i][0]; d[1] = (__bf16)ta[i][1];
      d[2] = (__bf16)ta[i][2]; d[3] = (__bf16)ta[i][3];
    }
#pragma unroll
    for (int i = 0; i < 2; ++i) {
      __bf16* d = sB + (smR + i * 32) * SB_STR + skk;
      d[0] = (__bf16)tb[i][0]; d[1] = (__bf16)tb[i][1];
      d[2] = (__bf16)tb[i][2]; d[3] = (__bf16)tb[i][3];
    }
    __syncthreads();
    v16bf af[2], bf[2];
    af[0] = frag_a(sA + (wm * 32 +  0) * SA_STR, SA_STR);
    af[1] = frag_a(sA + (wm * 32 + 16) * SA_STR, SA_STR);
    bf[0] = frag_b(sB + (wn * 32 +  0) * SB_STR, SB_STR);
    bf[1] = frag_b(sB + (wn * 32 + 16) * SB_STR, SB_STR);
    for (int i = 0; i < 2; ++i)
      for (int j = 0; j < 2; ++j) acc[i][j] = wmma_bf16(af[i], bf[j], acc[i][j]);
    __syncthreads();
  }

  for (int i = 0; i < 2; ++i)
    for (int j = 0; j < 2; ++j)
      for (int r = 0; r < 8; ++r) {
        int m = m0 + wm * 32 + i * 16 + r + ((lane >> 4) << 3);
        int n = n0 + wn * 32 + j * 16 + (lane & 15);
        float v = acc[i][j][r] + bias[n];
        if (mode == 0) {
          ((float*)C)[(size_t)m * ldc + n] = v;
        } else {
          int b = m >> 10, kk = m & 1023, h = n >> 6, d = n & 63;
          size_t idx = (mode == 1)
              ? ((size_t)(b * 16 + h) * 1024 + kk) * 64 + d
              : ((size_t)(b * 16 + h) * 64 + d) * 1024 + kk;
          ((__bf16*)C)[idx] = (__bf16)v;
        }
      }
}

// ---------------------------------------------------------------------------
// Fused per-(b,h) monotonic attention: phi & reg scores (WMMA) -> log-space
// DP scan -> softmax -> attn@V (WMMA).  64 blocks x 512 threads (16 waves).
// K tile (1024x64 bf16) staged to padded LDS by the Tensor Data Mover.
// ---------------------------------------------------------------------------
#define SXK_STR 72     /* bf16 row stride for K tile (64 data + 8 pad)   */
#define SS_STR  1028   /* f32 row stride for score tiles                 */
#define SP_STR  1032   /* bf16 row stride for probability tile           */
#define OFF_XK    0
#define OFF_Q     147456
#define OFF_SPHI  152064   /* aliased with sP (dead after DP)            */
#define OFF_SREG  217856
#define OFF_ALPHA 283648
#define SMEM_BYTES 287744

__global__ __launch_bounds__(512) void mono_attn_fused(
    const float* __restrict__ YQ, const __bf16* __restrict__ XKb,
    const __bf16* __restrict__ XVT, float* __restrict__ XO) {
  extern __shared__ char smem[];
  __bf16* sXK   = (__bf16*)(smem + OFF_XK);     // [1024][72]
  __bf16* sQ    = (__bf16*)(smem + OFF_Q);      // [2][16][72]
  float*  sSphi = (float*)(smem + OFF_SPHI);    // [16][1028]
  __bf16* sP    = (__bf16*)(smem + OFF_SPHI);   // [16][1032] (alias)
  float*  sSreg = (float*)(smem + OFF_SREG);    // [16][1028]
  float*  sAlpha= (float*)(smem + OFF_ALPHA);   // [1024]

  const int bh = blockIdx.x, b = bh >> 4, h = bh & 15;
  const int tid = threadIdx.x, lane = tid & 31, wave = tid >> 5;

  const __bf16* xk = XKb + (size_t)bh * 1024 * 64;

#if USE_TDM
  // Tensor Data Mover: 2D tile 64x1024 (bf16), LDS padding 4 DWORDs every
  // 32 DWORDs -> destination row stride 144B = 72 bf16 (matches SXK_STR).
  if (wave == 0) {
    unsigned lds_base =
        (unsigned)(size_t)(__attribute__((address_space(3))) char*)smem + OFF_XK;
    unsigned long long ga = (unsigned long long)(const void*)xk;
    u32x4 g0;
    g0[0] = 1u;                                   // count=1, user mode
    g0[1] = lds_base;                             // lds_addr
    g0[2] = (unsigned)(ga & 0xffffffffu);         // global_addr[95:64]
    g0[3] = (unsigned)((ga >> 32) & 0x01ffffffu)  // global_addr[120:96]
            | 0x80000000u;                        // type=2 ("image")
    u32x8 g1;
    g1[0] = (1u << 16)      /* data_size=2B */
          | (1u << 20)      /* pad_enable  */
          | (4u << 22)      /* pad_interval: 32 DWORDs */
          | (3u << 25);     /* pad_amount:   4 DWORDs  */
    g1[1] = 64u << 16;      // tensor_dim0 = 64 (bits 79:48)
    g1[2] = 1024u << 16;    // tensor_dim1 = 1024 (bits 111:80)
    g1[3] = 64u << 16;      // tile_dim0 = 64 (bits 127:112)
    g1[4] = 1024u;          // tile_dim1 = 1024 (bits 143:128)
    g1[5] = 64u;            // tensor_dim0_stride = 64 (bits 207:160)
    g1[6] = 0u;
    g1[7] = 0u;
    asm volatile("tensor_load_to_lds %0, %1" :: "s"(g0), "s"(g1) : "memory");
    __builtin_amdgcn_s_wait_tensorcnt(0);
  }
#else
  for (int c = tid; c < 1024 * 8; c += 512) {
    int k = c >> 3, dd = (c & 7) << 3;
    *(v8bf*)(sXK + k * SXK_STR + dd) = *(const v8bf*)(xk + k * 64 + dd);
  }
#endif
  for (int k = tid; k < 1024; k += 512) sAlpha[k] = (k == 0) ? 0.f : NEG_INF;
  __syncthreads();

  const __bf16* xvt = XVT + (size_t)bh * 64 * 1024;

  for (int q0 = 0; q0 < 512; q0 += 16) {
    // stage 16 query rows of xqa (mat 0) and xq (mat 1), f32 -> bf16
    {
      int c = tid;  // 512 chunks of 4 floats, one per thread
      int mat = c >> 8, rem = c & 255, qq = rem >> 4, dd = (rem & 15) << 2;
      v4f v = *(const v4f*)(YQ + (size_t)(b * 512 + q0 + qq) * 2048 +
                            mat * 1024 + h * 64 + dd);
      __bf16* d = sQ + (mat * 16 + qq) * SXK_STR + dd;
      d[0] = (__bf16)v[0]; d[1] = (__bf16)v[1]; d[2] = (__bf16)v[2]; d[3] = (__bf16)v[3];
    }
    __syncthreads();

    // score GEMMs: 2 matrices x 64 key-tiles = 128 WMMA jobs over 16 waves.
    // A-fragments (4 distinct: mat x d-chunk) hoisted out of the job loop.
    {
      v16bf afr[2][2];
#pragma unroll
      for (int mat = 0; mat < 2; ++mat)
#pragma unroll
        for (int dc = 0; dc < 2; ++dc)
          afr[mat][dc] = frag_a(sQ + mat * 16 * SXK_STR + dc * 32, SXK_STR);
#pragma unroll
      for (int i = 0; i < 8; ++i) {
        int job = wave + (i << 4);
        int mat = job >> 6, kt = job & 63;
        v8f acc; for (int r = 0; r < 8; ++r) acc[r] = 0.f;
#pragma unroll
        for (int dc = 0; dc < 2; ++dc) {
          v16bf bb = frag_b(sXK + (kt << 4) * SXK_STR + dc * 32, SXK_STR);
          acc = wmma_bf16(afr[mat][dc], bb, acc);
        }
        float* S = mat ? sSreg : sSphi;
#pragma unroll
        for (int r = 0; r < 8; ++r) {
          int qq = r + ((lane >> 4) << 3);
          int kk = (kt << 4) + (lane & 15);
          float v = acc[r];
          S[qq * SS_STR + kk] = mat ? v * SCALE_QK : tanhf(v) * CLAMP_PHI;
        }
      }
    }
    __syncthreads();

    // sequential DP over the 16 query columns of this block
    for (int qq = 0; qq < 16; ++qq) {
      float nv[2];
#pragma unroll
      for (int j = 0; j < 2; ++j) {
        int k = tid + (j << 9);
        float ak = sAlpha[k];
        float phik = sSphi[qq * SS_STR + k];
        sSreg[qq * SS_STR + k] += ak;                 // score += mono(log alpha)
        float stay = ak + log_sigmoid(phik);
        float move = NEG_INF;
        if (k > 0)
          move = sAlpha[k - 1] + log_sigmoid(-sSphi[qq * SS_STR + k - 1]);
        nv[j] = logaddexp_f(stay, move);
      }
      __syncthreads();
      sAlpha[tid] = nv[0];
      sAlpha[tid + 512] = nv[1];
      __syncthreads();
    }

    // softmax over keys: wave w owns row qq = w
    {
      float* row = sSreg + wave * SS_STR;
      float m = NEG_INF;
      for (int k = lane; k < 1024; k += 32) m = fmaxf(m, row[k]);
      for (int off = 16; off; off >>= 1) m = fmaxf(m, __shfl_xor(m, off, 32));
      float s = 0.f;
      for (int k = lane; k < 1024; k += 32) {
        float e = __expf(row[k] - m); row[k] = e; s += e;
      }
      for (int off = 16; off; off >>= 1) s += __shfl_xor(s, off, 32);
      float inv = 1.0f / s;
      for (int k = lane; k < 1024; k += 32)
        sP[wave * SP_STR + k] = (__bf16)(row[k] * inv);
    }
    __syncthreads();

    // attn @ V: waves 0..3 each own a 16-wide d-tile, K=1024 in 32 chunks
    if (wave < 4) {
      int n0 = wave << 4;
      v8f acc; for (int r = 0; r < 8; ++r) acc[r] = 0.f;
      for (int k0 = 0; k0 < 1024; k0 += 32) {
        if (k0 + 32 < 1024)
          __builtin_prefetch(xvt + (size_t)n0 * 1024 + k0 + 32, 0, 1);
        v16bf a = frag_a(sP + k0, SP_STR);
        v16bf bb = frag_b(xvt + (size_t)n0 * 1024 + k0, 1024);
        acc = wmma_bf16(a, bb, acc);
      }
#pragma unroll
      for (int r = 0; r < 8; ++r) {
        int qloc = r + ((lane >> 4) << 3);
        int dd = n0 + (lane & 15);
        XO[(size_t)(b * 512 + q0 + qloc) * 1024 + h * 64 + dd] = acc[r];
      }
    }
    __syncthreads();
  }
}

// ---------------------------------------------------------------------------
extern "C" void kernel_launch(void* const* d_in, const int* in_sizes, int n_in,
                              void* d_out, int out_size, void* d_ws, size_t ws_size,
                              hipStream_t stream) {
  (void)in_sizes; (void)n_in; (void)out_size; (void)ws_size;
  const float* query = (const float*)d_in[0];   // (4,512,1024)
  const float* key   = (const float*)d_in[1];   // (4,1024,1024)
  const float* value = (const float*)d_in[2];   // (4,1024,1024)
  const float* ipw   = (const float*)d_in[3];   // (4096,1024)
  const float* ipb   = (const float*)d_in[4];   // (4096)
  const float* outw  = (const float*)d_in[5];   // (1024,1024)
  const float* outb  = (const float*)d_in[6];   // (1024)
  float* out = (float*)d_out;                   // (4,512,1024)

  char* ws = (char*)d_ws;
  float*  YQ  = (float*)(ws);                    // 2048x2048 f32  (16 MB)
  __bf16* XKb = (__bf16*)(ws + 16777216);        // [64][1024][64] bf16 (8 MB)
  __bf16* XVT = (__bf16*)(ws + 25165824);        // [64][64][1024] bf16 (8 MB)
  float*  XO  = (float*)(ws + 33554432);         // 2048x1024 f32  (8 MB)

  // q-mono + q projections (W rows 0..2047) -> YQ
  mono_proj_gemm<<<dim3(32, 16), 256, 0, stream>>>(
      query, ipw, ipb, YQ, 2048, 2048, 1024, 0, 2048);
  // key projection (W rows 2048..3071) -> XKb bf16 [b][h][k][d]
  mono_proj_gemm<<<dim3(16, 32), 256, 0, stream>>>(
      key, ipw + (size_t)2048 * 1024, ipb + 2048, XKb, 4096, 1024, 1024, 1, 0);
  // value projection (W rows 3072..4095) -> XVT bf16 [b][h][d][k]
  mono_proj_gemm<<<dim3(16, 32), 256, 0, stream>>>(
      value, ipw + (size_t)3072 * 1024, ipb + 3072, XVT, 4096, 1024, 1024, 2, 0);
  // fused monotonic attention per (b,h)
  mono_attn_fused<<<64, 512, SMEM_BYTES, stream>>>(YQ, XKb, XVT, XO);
  // output projection -> d_out
  mono_proj_gemm<<<dim3(16, 16), 256, 0, stream>>>(
      XO, outw, outb, out, 2048, 1024, 1024, 0, 1024);
}
